// TrajectoryPrediction_16922171147077
// MI455X (gfx1250) — matrix operand order repaired and checked by reading, not compiled, not run on hardware
//
#include <hip/hip_runtime.h>

typedef _Float16 h16v __attribute__((ext_vector_type(16)));
typedef _Float16 h8v  __attribute__((ext_vector_type(8)));
typedef float    f8v  __attribute__((ext_vector_type(8)));

#define THREADS 256
#define GSZ 64
#define NPED 16384
#define PRED 12
#define TOFF 8          // T=20, last 12 steps

// --- shared param block offsets (floats) ---
#define P_H2GW  0     // (32,2)
#define P_H2GB  64
#define P_H2AW  66
#define P_H2AB  130
#define P_GATTW 132   // (2,32)
#define P_GATTB 196
#define P_AATTW 228
#define P_AATTB 292
#define P_ASRC1 324   // (4,16)
#define P_ADST1 388
#define P_BIAS1 452
#define P_ASRC2 468   // (32)
#define P_ADST2 500
#define P_BIAS2 532
#define P_GB    564   // g_bih+g_bhh + emb_b@Wih^T   (128)
#define P_AB    692
#define P_EWG   820   // goal_emb_W @ g_Wih^T  (2,128)
#define P_EWA   1076
#define P_TOT   1332

struct Ptrs {
  const float *goal_real, *action_real, *goal_h0, *action_h0;
  const float *goal_emb_W, *goal_emb_b, *act_emb_W, *act_emb_b;
  const float *g_Wih, *g_Whh, *g_bih, *g_bhh;
  const float *a_Wih, *a_Whh, *a_bih, *a_bhh;
  const float *h2g_W, *h2g_b, *h2a_W, *h2a_b;
  const float *gatt_W, *gatt_b, *aatt_W, *aatt_b;
  const float *w1, *asrc1, *adst1, *bias1;
  const float *w2, *asrc2, *adst2, *bias2;
  float* out;
};

// single-instruction TRANS32 ops (co-issue with WMMA on CDNA5)
__device__ __forceinline__ float frcp(float x)  { return __builtin_amdgcn_rcpf(x); }
__device__ __forceinline__ float frsq(float x)  { return __builtin_amdgcn_rsqf(x); }
__device__ __forceinline__ float sigm(float x)  { return frcp(1.f + __expf(-x)); }
// tanh = 1 - 2/(e^{2x}+1): exact at both saturation ends
__device__ __forceinline__ float tanh_f(float x){ return 1.f - 2.f * frcp(__expf(2.f * x) + 1.f); }

// A fragment (16x32 f16), row-major LDS matrix, stride ld halves (ld,k0 mult of 8).
// ISA 7.12.2: lane<16 -> M=lane, K={k0..k0+7, k0+16..k0+23}; lane>=16 adds +8 to K.
__device__ __forceinline__ h16v loadA(const _Float16* S, int ld, int m0, int k0, int lane) {
  int r  = m0 + (lane & 15);
  int hb = (lane >> 4) << 3;
  const _Float16* p = S + r * ld + k0 + hb;
  h8v lo = *(const h8v*)p;
  h8v hi = *(const h8v*)(p + 16);
  return __builtin_shufflevector(lo, hi, 0,1,2,3,4,5,6,7,8,9,10,11,12,13,14,15);
}

// B fragment (32x16 f16) from K-contiguous (transposed) storage BT[n][k], stride ld.
__device__ __forceinline__ h16v loadBT(const _Float16* S, int ld, int n0, int k0, int lane) {
  int n  = n0 + (lane & 15);
  int hb = (lane >> 4) << 3;
  const _Float16* p = S + n * ld + k0 + hb;
  h8v lo = *(const h8v*)p;
  h8v hi = *(const h8v*)(p + 16);
  return __builtin_shufflevector(lo, hi, 0,1,2,3,4,5,6,7,8,9,10,11,12,13,14,15);
}

__device__ __forceinline__ f8v wmma16(h16v a, h16v b, f8v c) {
  return __builtin_amdgcn_wmma_f32_16x16x32_f16(false, a, false, b, (short)0, c, false, false);
}

__global__ __launch_bounds__(THREADS) void traj_pred_kernel(Ptrs P) {
  // ---- LDS: one 64-ped group resident for all 12 steps (~157KB -> 2 blocks/WGP) ----
  __shared__ __align__(16) _Float16 WgT[128 * 32];     // Whh^T, K-contig [col][k]
  __shared__ __align__(16) _Float16 WaT[128 * 32];
  __shared__ __align__(16) _Float16 W1T[4 * 16 * 32];  // per-head [o][k]
  __shared__ __align__(16) _Float16 W2T[32 * 64];      // [o][k]
  __shared__ __align__(16) float gh[GSZ * 32], gc[GSZ * 32];
  __shared__ __align__(16) float ah[GSZ * 32], ac[GSZ * 32];
  __shared__ __align__(16) _Float16 ghh[GSZ * 32];     // f16 shadow of gh (LSTM A operand)
  __shared__ __align__(16) _Float16 ahh[GSZ * 32];     // f16 shadow of ah
  __shared__ __align__(16) _Float16 Af[GSZ * 64];      // normalized inputs f16
  __shared__ __align__(16) _Float16 Bf[4096];          // hp / hp2 transposed f16
  __shared__ __align__(16) float fscr[GSZ * 128];      // gates f32 / GAT1 output
  __shared__ __align__(16) _Float16 attn[4 * 64 * 72]; // attn probs, row stride 72
  __shared__ float sarr[256], darr[256];
  __shared__ float mArr[64], ivArr[64];
  __shared__ float gx[128], ax[128];                   // per-step raw (x0,x1) inputs
  __shared__ float prm[P_TOT];

  const int tid  = threadIdx.x;
  const int lane = tid & 31;
  const int wid  = tid >> 5;
  const int nbase = blockIdx.x * GSZ;

  // ================= init stage 1: params, weights, state =================
  {
    const float* srcs[14] = {P.h2g_W, P.h2g_b, P.h2a_W, P.h2a_b,
                             P.gatt_W, P.gatt_b, P.aatt_W, P.aatt_b,
                             P.asrc1, P.adst1, P.bias1, P.asrc2, P.adst2, P.bias2};
    const int offs[14] = {P_H2GW,P_H2GB,P_H2AW,P_H2AB,P_GATTW,P_GATTB,P_AATTW,P_AATTB,
                          P_ASRC1,P_ADST1,P_BIAS1,P_ASRC2,P_ADST2,P_BIAS2};
    const int lens[14] = {64,2,64,2,64,32,64,32,64,64,16,32,32,32};
    for (int a = 0; a < 14; ++a)
      for (int i = tid; i < lens[a]; i += THREADS) prm[offs[a] + i] = srcs[a][i];
    for (int i = tid; i < 128; i += THREADS) {
      prm[P_GB + i] = P.g_bih[i] + P.g_bhh[i];
      prm[P_AB + i] = P.a_bih[i] + P.a_bhh[i];
    }
  }
  // Whh^T K-contiguous: BT[col][k] = Whh[col][k] == row-major g_Whh directly
  for (int i = tid; i < 128 * 32; i += THREADS) {
    WgT[i] = (_Float16)P.g_Whh[i];
    WaT[i] = (_Float16)P.a_Whh[i];
  }
  for (int i = tid; i < 4 * 32 * 16; i += THREADS) {
    int h = i >> 9, rem = i & 511, o = rem >> 5, kk = rem & 31;
    W1T[i] = (_Float16)P.w1[h * 512 + kk * 16 + o];
  }
  for (int i = tid; i < 64 * 32; i += THREADS) {
    int o = i >> 6, kk = i & 63;
    W2T[i] = (_Float16)P.w2[kk * 32 + o];
  }
  for (int i = tid; i < GSZ * 32; i += THREADS) {
    int n = i >> 5, j = i & 31;
    float g0 = P.goal_h0[(nbase + n) * 32 + j];
    float a0 = P.action_h0[(nbase + n) * 32 + j];
    gh[i] = g0;  ah[i] = a0;  ghh[i] = (_Float16)g0;  ahh[i] = (_Float16)a0;
    gc[i] = 0.f; ac[i] = 0.f;
  }
  __syncthreads();
  // ===== init stage 2: fold rank-2 embedding: EW = W_emb @ Wih^T, bias += b_emb @ Wih^T
  for (int i = tid; i < 256; i += THREADS) {
    int r = i >> 7, col = i & 127;
    float sg = 0.f, sa = 0.f;
    for (int j = 0; j < 16; ++j) {
      sg += P.goal_emb_W[r * 16 + j] * P.g_Wih[col * 16 + j];
      sa += P.act_emb_W[r * 16 + j]  * P.a_Wih[col * 16 + j];
    }
    prm[P_EWG + i] = sg;  prm[P_EWA + i] = sa;
    if (r == 0) {
      float bg = 0.f, ba = 0.f;
      for (int j = 0; j < 16; ++j) {
        bg += P.goal_emb_b[j] * P.g_Wih[col * 16 + j];
        ba += P.act_emb_b[j]  * P.a_Wih[col * 16 + j];
      }
      prm[P_GB + col] += bg;
      prm[P_AB + col] += ba;
    }
  }
  __syncthreads();

  for (int s = 0; s < PRED; ++s) {
    float go0 = 0.f, go1 = 0.f;   // goal_out lives in registers of threads 0..63

    // ============ GOAL gates: gh(64x32) @ Whh^T -> fscr; also fetch step inputs ============
    {
      if (tid < 128)
        gx[tid] = P.goal_real[((size_t)(TOFF + s) * NPED + nbase) * 2 + tid];
      else
        ax[tid - 128] = P.action_real[((size_t)(TOFF + s) * NPED + nbase) * 2 + (tid - 128)];
      int n0 = wid * 16;
      for (int m = 0; m < 4; ++m) {
        f8v c = {};
        c = wmma16(loadA(ghh, 32, m * 16, 0, lane), loadBT(WgT, 32, n0, 0, lane), c);
        int col = n0 + (lane & 15), rb = m * 16 + ((lane >> 4) << 3);
        for (int i = 0; i < 8; ++i) fscr[(rb + i) * 128 + col] = c[i];
      }
    }
    __syncthreads();
    for (int q = tid; q < GSZ * 32; q += THREADS) {   // LSTM cell + rank-2 input term
      int n = q >> 5, j = q & 31;
      const float* g = fscr + n * 128;
      float x0 = gx[n * 2], x1 = gx[n * 2 + 1];
      float ig = sigm  (g[j]      + x0*prm[P_EWG+j]      + x1*prm[P_EWG+128+j] + prm[P_GB+j]);
      float fg = sigm  (g[32 + j] + x0*prm[P_EWG+32+j]   + x1*prm[P_EWG+160+j] + prm[P_GB+32+j]);
      float gg = tanh_f(g[64 + j] + x0*prm[P_EWG+64+j]   + x1*prm[P_EWG+192+j] + prm[P_GB+64+j]);
      float og = sigm  (g[96 + j] + x0*prm[P_EWG+96+j]   + x1*prm[P_EWG+224+j] + prm[P_GB+96+j]);
      float c2 = fg * gc[q] + ig * gg;
      gc[q] = c2;  gh[q] = og * tanh_f(c2);
    }
    __syncthreads();
    if (tid < GSZ) {   // goal_out -> global + registers
      int n = tid;
      go0 = prm[P_H2GB + 0];  go1 = prm[P_H2GB + 1];
      for (int j = 0; j < 32; ++j) {
        float hv = gh[n * 32 + j];
        go0 += hv * prm[P_H2GW + j * 2 + 0];
        go1 += hv * prm[P_H2GW + j * 2 + 1];
      }
      float* op = P.out + ((size_t)(0 * PRED + s) * NPED + nbase + n) * 2;
      op[0] = go0;  op[1] = go1;
    }

    // ============ ACTION gates (independent of goal side) ============
    {
      int n0 = wid * 16;
      for (int m = 0; m < 4; ++m) {
        f8v c = {};
        c = wmma16(loadA(ahh, 32, m * 16, 0, lane), loadBT(WaT, 32, n0, 0, lane), c);
        int col = n0 + (lane & 15), rb = m * 16 + ((lane >> 4) << 3);
        for (int i = 0; i < 8; ++i) fscr[(rb + i) * 128 + col] = c[i];
      }
    }
    __syncthreads();
    for (int q = tid; q < GSZ * 32; q += THREADS) {
      int n = q >> 5, j = q & 31;
      const float* g = fscr + n * 128;
      float x0 = ax[n * 2], x1 = ax[n * 2 + 1];
      float ig = sigm  (g[j]      + x0*prm[P_EWA+j]      + x1*prm[P_EWA+128+j] + prm[P_AB+j]);
      float fg = sigm  (g[32 + j] + x0*prm[P_EWA+32+j]   + x1*prm[P_EWA+160+j] + prm[P_AB+32+j]);
      float gg = tanh_f(g[64 + j] + x0*prm[P_EWA+64+j]   + x1*prm[P_EWA+192+j] + prm[P_AB+64+j]);
      float og = sigm  (g[96 + j] + x0*prm[P_EWA+96+j]   + x1*prm[P_EWA+224+j] + prm[P_AB+96+j]);
      float c2 = fg * ac[q] + ig * gg;
      ac[q] = c2;  ah[q] = og * tanh_f(c2);
    }
    __syncthreads();
    if (tid < GSZ) {   // goal-attention gate on ah
      int n = tid;
      float lg[32], mx = -3.4e38f;
      for (int c = 0; c < 32; ++c) {
        float v = go0 * prm[P_GATTW + c] + go1 * prm[P_GATTW + 32 + c] + prm[P_GATTB + c];
        lg[c] = v;  mx = fmaxf(mx, v);
      }
      float sum = 0.f;
      for (int c = 0; c < 32; ++c) { lg[c] = __expf(lg[c] - mx); sum += lg[c]; }
      float inv = frcp(sum);
      for (int c = 0; c < 32; ++c) ah[n * 32 + c] *= lg[c] * inv;
    }
    __syncthreads();

    // ======================= GAT LAYER 1 =======================
    {   // instance norm partials: 8 threads per feature
      int f = tid & 31, part = tid >> 5;
      float sm = 0.f, sq = 0.f;
      for (int n = part * 8; n < part * 8 + 8; ++n) { float v = ah[n * 32 + f]; sm += v; sq += v * v; }
      sarr[tid] = sm;  darr[tid] = sq;
    }
    __syncthreads();
    if (tid < 32) {
      float sm = 0.f, sq = 0.f;
      for (int p = 0; p < 8; ++p) { sm += sarr[p * 32 + tid]; sq += darr[p * 32 + tid]; }
      float m = sm * (1.f / GSZ);
      mArr[tid] = m;  ivArr[tid] = frsq(sq * (1.f / GSZ) - m * m + 1e-5f);
    }
    __syncthreads();
    for (int q = tid; q < GSZ * 32; q += THREADS) {
      int f = q & 31;
      Af[q] = (_Float16)((ah[q] - mArr[f]) * ivArr[f]);
    }
    __syncthreads();
    for (int j = wid; j < 16; j += 8) {   // hp = xn @ w1[h] -> Bf transposed [h][o][node]
      int h = j >> 2, m = j & 3;
      f8v c = {};
      c = wmma16(loadA(Af, 32, m * 16, 0, lane), loadBT(W1T + h * 512, 32, 0, 0, lane), c);
      int o = lane & 15, nb2 = m * 16 + ((lane >> 4) << 3);
      for (int i = 0; i < 8; ++i) Bf[h * 1024 + o * 64 + nb2 + i] = (_Float16)c[i];
    }
    __syncthreads();
    {   // s,d per (head,node)
      int h = tid >> 6, n = tid & 63;
      float sv = 0.f, dv = 0.f;
      for (int o = 0; o < 16; ++o) {
        float hp = (float)Bf[h * 1024 + o * 64 + n];
        sv += hp * prm[P_ASRC1 + h * 16 + o];
        dv += hp * prm[P_ADST1 + h * 16 + o];
      }
      sarr[tid] = sv;  darr[tid] = dv;
    }
    __syncthreads();
    {   // leaky-relu + row softmax -> attn f16
      int h = tid >> 6, r = tid & 63;
      float sv = sarr[tid];
      const float* dp = darr + h * 64;
      _Float16* row = attn + h * 4608 + r * 72;
      float mx = -3.4e38f;
      for (int c = 0; c < 64; ++c) {
        float v = sv + dp[c];  v = v > 0.f ? v : 0.2f * v;
        mx = fmaxf(mx, v);
      }
      float sum = 0.f;
      for (int c = 0; c < 64; ++c) {
        float v = sv + dp[c];  v = v > 0.f ? v : 0.2f * v;
        float e = __expf(v - mx);
        row[c] = (_Float16)e;  sum += e;
      }
      float inv = frcp(sum);
      for (int c = 0; c < 64; ++c) row[c] = (_Float16)((float)row[c] * inv);
    }
    __syncthreads();
    for (int j = wid; j < 16; j += 8) {   // out1 = attn @ hp + bias1, ELU -> fscr (ld 64)
      int h = j >> 2, m = j & 3;
      f8v c = {};
      for (int kc = 0; kc < 2; ++kc)
        c = wmma16(loadA(attn + h * 4608, 72, m * 16, kc * 32, lane),
                   loadBT(Bf + h * 1024, 64, 0, kc * 32, lane), c);
      int o = lane & 15, nb2 = m * 16 + ((lane >> 4) << 3);
      for (int i = 0; i < 8; ++i) {
        float x = c[i] + prm[P_BIAS1 + o];
        x = x > 0.f ? x : (__expf(x) - 1.f);
        fscr[(nb2 + i) * 64 + h * 16 + o] = x;
      }
    }
    __syncthreads();

    // ======================= GAT LAYER 2 =======================
    {   // instance norm partials: 4 threads per feature (64 features)
      int f = tid & 63, part = tid >> 6;
      float sm = 0.f, sq = 0.f;
      for (int n = part * 16; n < part * 16 + 16; ++n) { float v = fscr[n * 64 + f]; sm += v; sq += v * v; }
      sarr[tid] = sm;  darr[tid] = sq;
    }
    __syncthreads();
    if (tid < 64) {
      float sm = 0.f, sq = 0.f;
      for (int p = 0; p < 4; ++p) { sm += sarr[p * 64 + tid]; sq += darr[p * 64 + tid]; }
      float m = sm * (1.f / GSZ);
      mArr[tid] = m;  ivArr[tid] = frsq(sq * (1.f / GSZ) - m * m + 1e-5f);
    }
    __syncthreads();
    for (int q = tid; q < GSZ * 64; q += THREADS) {
      int f = q & 63;
      Af[q] = (_Float16)((fscr[q] - mArr[f]) * ivArr[f]);
    }
    __syncthreads();
    {   // hp2 = xn2 @ w2 -> Bf transposed [o][node]
      int m = wid >> 1, nt = wid & 1;
      f8v c = {};
      for (int kc = 0; kc < 2; ++kc)
        c = wmma16(loadA(Af, 64, m * 16, kc * 32, lane),
                   loadBT(W2T, 64, nt * 16, kc * 32, lane), c);
      int o = nt * 16 + (lane & 15), nb2 = m * 16 + ((lane >> 4) << 3);
      for (int i = 0; i < 8; ++i) Bf[o * 64 + nb2 + i] = (_Float16)c[i];
    }
    __syncthreads();
    if (tid < 64) {   // s2,d2
      int n = tid;  float sv = 0.f, dv = 0.f;
      for (int o = 0; o < 32; ++o) {
        float hp = (float)Bf[o * 64 + n];
        sv += hp * prm[P_ASRC2 + o];
        dv += hp * prm[P_ADST2 + o];
      }
      sarr[n] = sv;  darr[n] = dv;
    }
    __syncthreads();
    if (tid < 64) {   // row softmax (head-0 area of attn)
      int r = tid;  float sv = sarr[r];
      _Float16* row = attn + r * 72;
      float mx = -3.4e38f;
      for (int c = 0; c < 64; ++c) {
        float v = sv + darr[c];  v = v > 0.f ? v : 0.2f * v;
        mx = fmaxf(mx, v);
      }
      float sum = 0.f;
      for (int c = 0; c < 64; ++c) {
        float v = sv + darr[c];  v = v > 0.f ? v : 0.2f * v;
        float e = __expf(v - mx);
        row[c] = (_Float16)e;  sum += e;
      }
      float inv = frcp(sum);
      for (int c = 0; c < 64; ++c) row[c] = (_Float16)((float)row[c] * inv);
    }
    __syncthreads();
    {   // ah = attn2 @ hp2 + bias2  (also refresh f16 shadow)
      int m = wid >> 1, nt = wid & 1;
      f8v c = {};
      for (int kc = 0; kc < 2; ++kc)
        c = wmma16(loadA(attn, 72, m * 16, kc * 32, lane),
                   loadBT(Bf, 64, nt * 16, kc * 32, lane), c);
      int o = nt * 16 + (lane & 15), nb2 = m * 16 + ((lane >> 4) << 3);
      for (int i = 0; i < 8; ++i) {
        float v = c[i] + prm[P_BIAS2 + o];
        ah[(nb2 + i) * 32 + o] = v;
        ahh[(nb2 + i) * 32 + o] = (_Float16)v;
      }
    }
    __syncthreads();

    // ===== act_out -> global; action-attention gate on gh (+ f16 shadow refresh) =====
    if (tid < GSZ) {
      int n = tid;
      float ao0 = prm[P_H2AB + 0], ao1 = prm[P_H2AB + 1];
      for (int j = 0; j < 32; ++j) {
        float hv = ah[n * 32 + j];
        ao0 += hv * prm[P_H2AW + j * 2 + 0];
        ao1 += hv * prm[P_H2AW + j * 2 + 1];
      }
      float* op = P.out + ((size_t)(1 * PRED + s) * NPED + nbase + n) * 2;
      op[0] = ao0;  op[1] = ao1;
      float lg[32], mx = -3.4e38f;
      for (int c = 0; c < 32; ++c) {
        float v = ao0 * prm[P_AATTW + c] + ao1 * prm[P_AATTW + 32 + c] + prm[P_AATTB + c];
        lg[c] = v;  mx = fmaxf(mx, v);
      }
      float sum = 0.f;
      for (int c = 0; c < 32; ++c) { lg[c] = __expf(lg[c] - mx); sum += lg[c]; }
      float inv = frcp(sum);
      for (int c = 0; c < 32; ++c) {
        float v = gh[n * 32 + c] * lg[c] * inv;
        gh[n * 32 + c] = v;
        ghh[n * 32 + c] = (_Float16)v;
      }
    }
    __syncthreads();
  }
}

extern "C" void kernel_launch(void* const* d_in, const int* in_sizes, int n_in,
                              void* d_out, int out_size, void* d_ws, size_t ws_size,
                              hipStream_t stream) {
  (void)in_sizes; (void)n_in; (void)out_size; (void)d_ws; (void)ws_size;
  Ptrs P;
  P.goal_real   = (const float*)d_in[0];
  P.action_real = (const float*)d_in[1];
  P.goal_h0     = (const float*)d_in[2];
  P.action_h0   = (const float*)d_in[3];
  P.goal_emb_W  = (const float*)d_in[4];
  P.goal_emb_b  = (const float*)d_in[5];
  P.act_emb_W   = (const float*)d_in[6];
  P.act_emb_b   = (const float*)d_in[7];
  P.g_Wih = (const float*)d_in[8];   P.g_Whh = (const float*)d_in[9];
  P.g_bih = (const float*)d_in[10];  P.g_bhh = (const float*)d_in[11];
  P.a_Wih = (const float*)d_in[12];  P.a_Whh = (const float*)d_in[13];
  P.a_bih = (const float*)d_in[14];  P.a_bhh = (const float*)d_in[15];
  P.h2g_W = (const float*)d_in[16];  P.h2g_b = (const float*)d_in[17];
  P.h2a_W = (const float*)d_in[18];  P.h2a_b = (const float*)d_in[19];
  P.gatt_W = (const float*)d_in[20]; P.gatt_b = (const float*)d_in[21];
  P.aatt_W = (const float*)d_in[22]; P.aatt_b = (const float*)d_in[23];
  P.w1 = (const float*)d_in[24];     P.asrc1 = (const float*)d_in[25];
  P.adst1 = (const float*)d_in[26];  P.bias1 = (const float*)d_in[27];
  P.w2 = (const float*)d_in[28];     P.asrc2 = (const float*)d_in[29];
  P.adst2 = (const float*)d_in[30];  P.bias2 = (const float*)d_in[31];
  P.out = (float*)d_out;
  hipLaunchKernelGGL(traj_pred_kernel, dim3(NPED / GSZ), dim3(THREADS), 0, stream, P);
}
